// HNet_9998683865089
// MI455X (gfx1250) — matrix operand order repaired and checked by reading, not compiled, not run on hardware
//
// MI455X / gfx1250 forward pass for the hierarchical patch transformer.
// Strategy: every linear layer and both attention GEMMs go through
// v_wmma_f32_16x16x32_bf16 (bf16 operands, f32 accumulate) -- the chip is
// matrix-core bound here (~50 GFLOP vs ~15us of HBM traffic at 23.3 TB/s).
// wave32 everywhere; LDS staging for GEMM tiles (via GLOBAL_LOAD_ASYNC_TO_LDS
// when the toolchain exposes it) and LDS P-fragment transpose in attention.

#include <hip/hip_runtime.h>
#include <cstdint>
#include <cstddef>

typedef __attribute__((ext_vector_type(16))) __bf16 v16bf;
typedef __attribute__((ext_vector_type(8)))  float  v8f;
typedef __attribute__((ext_vector_type(4)))  int    v4i;

constexpr int Bc = 128, Tc = 256, D1c = 64, K1c = 64, K2c = 16, NHc = 4, Vc = 50257;

// ---- gfx1250 async global->LDS path (ASYNCcnt), guarded so absence of the
// builtins falls back to the proven load+ds_store path. Device pass only.
// Probe result: builtin exists with signature (v4i AS1*, v4i AS3*, Ii, Ii).
#if defined(__gfx1250__) && __has_builtin(__builtin_amdgcn_global_load_async_to_lds_b128) && \
    __has_builtin(__builtin_amdgcn_s_wait_asynccnt)
#define USE_ASYNC_LDS 1
#else
#define USE_ASYNC_LDS 0
#endif

#if USE_ASYNC_LDS
__device__ __forceinline__ void async_b128(void* lds, const void* g) {
  // flat LDS pointers carry the LDS byte offset in their low 32 bits
  __builtin_amdgcn_global_load_async_to_lds_b128(
      (__attribute__((address_space(1))) v4i*)(uintptr_t)g,
      (__attribute__((address_space(3))) v4i*)(uint32_t)(uintptr_t)lds,
      0, 0);
}
__device__ __forceinline__ void async_wait() { __builtin_amdgcn_s_wait_asynccnt(0); }
#endif

__device__ __forceinline__ float gelu_f(float x) {
  float x3 = x * x * x;
  return 0.5f * x * (1.0f + tanhf(0.7978845608028654f * (x + 0.044715f * x3)));
}

// ---------------- embedding: x = tok_emb[idx] + pos_emb ----------------
__global__ void k_embed(const int* __restrict__ idx, const float* __restrict__ tok,
                        const float* __restrict__ pos, float* __restrict__ x, int T, int d) {
  int r = blockIdx.x;          // b*T + t
  int t = r % T;
  int token = idx[r];
  const float* tp = tok + (size_t)token * d;
  const float* pp = pos + (size_t)t * d;
  float* xp = x + (size_t)r * d;
  for (int c0 = threadIdx.x; c0 < d; c0 += blockDim.x) xp[c0] = tp[c0] + pp[c0];
}

// ---------------- layernorm ----------------
__global__ __launch_bounds__(128) void k_ln(const float* __restrict__ x, const float* __restrict__ g,
                                            float* __restrict__ y, int M, int d) {
  int row = blockIdx.x * 4 + (threadIdx.x >> 5);
  int lane = threadIdx.x & 31;
  if (row >= M) return;
  const float* xr = x + (size_t)row * d;
  float s = 0.f, s2 = 0.f;
  for (int i = lane; i < d; i += 32) { float v = xr[i]; s += v; s2 += v * v; }
  for (int o = 16; o; o >>= 1) { s += __shfl_xor(s, o, 32); s2 += __shfl_xor(s2, o, 32); }
  float mu = s / d;
  float var = s2 / d - mu * mu;
  float r = rsqrtf(var + 1e-5f);
  float* yr = y + (size_t)row * d;
  for (int i = lane; i < d; i += 32) yr[i] = (xr[i] - mu) * r * g[i];
}

// ---------------- WMMA GEMM: Y[M,N] = act(A[M,K] @ W[N,K]^T + bias) + Res ----------------
// 128 threads = 4 waves; block tile 32x64; each wave owns a 16x32 slab
// (two 16x16 WMMA accumulators sharing one A fragment).
__global__ __launch_bounds__(128) void k_gemm(const float* __restrict__ A, long lda,
                                              const float* __restrict__ W,
                                              const float* __restrict__ bias,
                                              const float* __restrict__ Res,
                                              float* __restrict__ Y,
                                              int M, int N, int K, int act) {
  __builtin_amdgcn_s_wait_tensorcnt(0);   // CDNA5 split-counter path (no TDM in flight -> free)
  __shared__ float As[32][36];            // 36-stride: every 16B row chunk aligned for B128
  __shared__ float Ws[64][36];
  int tid = threadIdx.x;
  int wave = tid >> 5, lane = tid & 31;
  int wm = wave >> 1, wn = wave & 1;
  int bm = blockIdx.y * 32, bn = blockIdx.x * 64;
  int hl = lane >> 4, l16 = lane & 15;
  int col0 = bn + wn * 32 + l16;
  int col1 = col0 + 16;
  float bv0 = (bias != nullptr && col0 < N) ? bias[col0] : 0.f;
  float bv1 = (bias != nullptr && col1 < N) ? bias[col1] : 0.f;
  v8f c0, c1;
#pragma unroll
  for (int j = 0; j < 8; j++) { c0[j] = bv0; c1[j] = bv1; }

  for (int k0 = 0; k0 < K; k0 += 32) {
#if USE_ASYNC_LDS
    bool fullA = (bm + 32 <= M) && (k0 + 32 <= K);
    bool fullW = (bn + 64 <= N) && (k0 + 32 <= K);
    if (fullA) {
      // 32x32 f32 tile = 256 B128 lane-transfers, DMA'd straight into LDS
      for (int i = tid; i < 256; i += 128) {
        int r = i >> 3, c4 = (i & 7) << 2;
        async_b128(&As[r][c4], &A[(size_t)(bm + r) * lda + k0 + c4]);
      }
    } else
#endif
    {
      for (int i = tid; i < 32 * 32; i += 128) {
        int r = i >> 5, cc = i & 31;
        int gm = bm + r, gk = k0 + cc;
        As[r][cc] = (gm < M && gk < K) ? A[(size_t)gm * lda + gk] : 0.f;
      }
    }
#if USE_ASYNC_LDS
    if (fullW) {
      for (int i = tid; i < 512; i += 128) {
        int r = i >> 3, c4 = (i & 7) << 2;
        async_b128(&Ws[r][c4], &W[(size_t)(bn + r) * K + k0 + c4]);
      }
    } else
#endif
    {
      if (k0 + 32 < K) __builtin_prefetch(&W[(size_t)(bn + (tid & 31)) * K + k0 + 32], 0, 0);
      for (int i = tid; i < 64 * 32; i += 128) {
        int r = i >> 5, cc = i & 31;
        int gn = bn + r, gk = k0 + cc;
        Ws[r][cc] = (gn < N && gk < K) ? W[(size_t)gn * K + gk] : 0.f;
      }
    }
#if USE_ASYNC_LDS
    async_wait();   // this wave's async stores to LDS complete
#endif
    __syncthreads();
    // Build bf16 fragments in contiguous k order, same for A and B (K 0-15 / 16-31
    // per lane-half).
    v16bf af, b0, b1;
#pragma unroll
    for (int e = 0; e < 16; e++) {
      int da = e + 16 * hl;
      int db = e + 16 * hl;
      af[e] = (__bf16)As[wm * 16 + l16][da];
      b0[e] = (__bf16)Ws[wn * 32 + l16][db];
      b1[e] = (__bf16)Ws[wn * 32 + 16 + l16][db];
    }
    c0 = __builtin_amdgcn_wmma_f32_16x16x32_bf16(false, af, false, b0, (short)0, c0, false, false);
    c1 = __builtin_amdgcn_wmma_f32_16x16x32_bf16(false, af, false, b1, (short)0, c1, false, false);
    __syncthreads();
  }
#pragma unroll
  for (int j = 0; j < 8; j++) {
    int row = bm + wm * 16 + j + 8 * hl;
    if (row < M) {
      if (col0 < N) {
        float v = c0[j];
        if (act == 1) v = gelu_f(v);
        if (Res != nullptr) v += Res[(size_t)row * N + col0];
        Y[(size_t)row * N + col0] = v;
      }
      if (col1 < N) {
        float v = c1[j];
        if (act == 1) v = gelu_f(v);
        if (Res != nullptr) v += Res[(size_t)row * N + col1];
        Y[(size_t)row * N + col1] = v;
      }
    }
  }
}

// ---------------- fused flash attention (one wave = 16 query rows of one head) ----------------
template <int HD>
__global__ __launch_bounds__(32) void k_attn(const float* __restrict__ Q, int ldq,
                                             const float* __restrict__ Kb, int ldk,
                                             const float* __restrict__ Vb, int ldv,
                                             float* __restrict__ O, int ldo,
                                             int S, int causal) {
  constexpr int NC = (HD + 31) / 32;  // K-chunks for Q.K^T (head dim padded to 32)
  constexpr int ND = HD / 16;         // 16-wide output column chunks
  __shared__ float Pb[16 * 16];
  int lane = threadIdx.x;
  int hl = lane >> 4, l16 = lane & 15;
  int qt = blockIdx.x, h = blockIdx.y, b = blockIdx.z;
  float scale = rsqrtf((float)HD);

  int qrow = qt * 16 + l16;
  const float* qp = Q + ((size_t)b * S + qrow) * ldq + h * HD;
  v16bf qa[NC];
#pragma unroll
  for (int cc = 0; cc < NC; cc++) {
#pragma unroll
    for (int e = 0; e < 16; e++) {
      int dd = 32 * cc + e + 16 * hl;
      qa[cc][e] = (dd < HD) ? (__bf16)(qp[dd] * scale) : (__bf16)0.f;
    }
  }

  float mr[8], lr[8];
  v8f acc[ND];
#pragma unroll
  for (int j = 0; j < 8; j++) { mr[j] = -3.0e38f; lr[j] = 0.f; }
#pragma unroll
  for (int dch = 0; dch < ND; dch++) {
#pragma unroll
    for (int j = 0; j < 8; j++) acc[dch][j] = 0.f;
  }

  int ktmax = causal ? qt : (S / 16 - 1);
  for (int kt = 0; kt <= ktmax; kt++) {
    v8f s;
#pragma unroll
    for (int j = 0; j < 8; j++) s[j] = 0.f;
    const float* kp = Kb + ((size_t)b * S + kt * 16 + l16) * ldk + h * HD;
#pragma unroll
    for (int cc = 0; cc < NC; cc++) {
      v16bf kf;
#pragma unroll
      for (int e = 0; e < 16; e++) {
        int dd = 32 * cc + e + 16 * hl;
        kf[e] = (dd < HD) ? (__bf16)kp[dd] : (__bf16)0.f;
      }
      s = __builtin_amdgcn_wmma_f32_16x16x32_bf16(false, qa[cc], false, kf, (short)0, s, false, false);
    }
    // online softmax: each C-fragment row lives in one VGPR across a 16-lane half
    float p8[8];
#pragma unroll
    for (int j = 0; j < 8; j++) {
      int qi = qt * 16 + j + 8 * hl;
      int key = kt * 16 + l16;
      float sv = s[j];
      if (causal && key > qi) sv = -1.0e30f;
      float tmax = sv;
      for (int o = 8; o; o >>= 1) tmax = fmaxf(tmax, __shfl_xor(tmax, o, 32));
      float nm = fmaxf(mr[j], tmax);
      float p = __expf(sv - nm);
      float ts = p;
      for (int o = 8; o; o >>= 1) ts += __shfl_xor(ts, o, 32);
      float alpha = __expf(mr[j] - nm);
      lr[j] = lr[j] * alpha + ts;
      mr[j] = nm;
#pragma unroll
      for (int dch = 0; dch < ND; dch++) acc[dch][j] *= alpha;
      p8[j] = p;
    }
    // transpose P (C layout -> A layout) through LDS
    __syncthreads();
#pragma unroll
    for (int j = 0; j < 8; j++) Pb[(j + 8 * hl) * 16 + l16] = p8[j];
    __syncthreads();
    v16bf pa;
#pragma unroll
    for (int e = 0; e < 16; e++) {
      int kk = e + 16 * hl;  // K 16..31 padded to zero
      pa[e] = (kk < 16) ? (__bf16)Pb[l16 * 16 + kk] : (__bf16)0.f;
    }
#pragma unroll
    for (int dch = 0; dch < ND; dch++) {
      v16bf vf;
#pragma unroll
      for (int e = 0; e < 16; e++) {
        int kk = e + 16 * hl;
        vf[e] = (kk < 16)
                    ? (__bf16)Vb[((size_t)b * S + kt * 16 + kk) * ldv + h * HD + dch * 16 + l16]
                    : (__bf16)0.f;
      }
      acc[dch] = __builtin_amdgcn_wmma_f32_16x16x32_bf16(false, pa, false, vf, (short)0, acc[dch], false, false);
    }
  }
#pragma unroll
  for (int dch = 0; dch < ND; dch++) {
#pragma unroll
    for (int j = 0; j < 8; j++) {
      int row = qt * 16 + j + 8 * hl;
      int colo = h * HD + dch * 16 + l16;
      O[((size_t)b * S + row) * ldo + colo] = acc[dch][j] / lr[j];
    }
  }
}

// ---------------- gate MLP (N=1, VALU) ----------------
__global__ __launch_bounds__(32) void k_gate(const float* __restrict__ x,
                                             const float* __restrict__ w1, const float* __restrict__ b1,
                                             const float* __restrict__ w2, const float* __restrict__ b2,
                                             float* __restrict__ gate, int d, int dh) {
  int row = blockIdx.x;
  int lane = threadIdx.x;
  const float* xr = x + (size_t)row * d;
  float acc = 0.f;
  if (lane < dh) {
    acc = b1[lane];
    const float* wr = w1 + (size_t)lane * d;
    for (int kk = 0; kk < d; kk++) acc += xr[kk] * wr[kk];
  }
  acc = fmaxf(acc, 0.f);
  float t = (lane < dh) ? acc * w2[lane] : 0.f;
  for (int o = 16; o; o >>= 1) t += __shfl_xor(t, o, 32);
  if (lane == 0) gate[row] = t + b2[0];
}

// ---------------- exact top-k (value desc, tie -> lower index, like lax.top_k) ----------------
__global__ __launch_bounds__(32) void k_topk(const float* __restrict__ gate, int* __restrict__ outidx,
                                             int T, int Ksel) {
  int b = blockIdx.x;
  int lane = threadIdx.x;
  const float* g = gate + (size_t)b * T;
  float v[8];
  int cnt = T >> 5;  // T in {64, 256}
  for (int i = 0; i < cnt; i++) v[i] = g[lane + 32 * i];
  unsigned taken = 0;
  for (int ssel = 0; ssel < Ksel; ssel++) {
    float bv = -3.4e38f;
    int bi = 0x7fffffff;
    for (int i = 0; i < cnt; i++) {
      if (!((taken >> i) & 1u)) {
        float vv = v[i];
        int ii = lane + 32 * i;
        if (vv > bv || (vv == bv && ii < bi)) { bv = vv; bi = ii; }
      }
    }
    for (int o = 16; o; o >>= 1) {
      float ov = __shfl_xor(bv, o, 32);
      int oi = __shfl_xor(bi, o, 32);
      if (ov > bv || (ov == bv && oi < bi)) { bv = ov; bi = oi; }
    }
    if (lane == 0) outidx[(size_t)b * Ksel + ssel] = bi;
    if ((bi & 31) == lane) taken |= 1u << (bi >> 5);
  }
}

// ---------------- small data-movement kernels ----------------
__global__ void k_gather_rows(const float* __restrict__ src, const int* __restrict__ idx,
                              float* __restrict__ dst, int Ssrc, int Ksel, int d) {
  int r = blockIdx.x;
  int b = r / Ksel;
  int t = idx[r];
  const float* sp = src + ((size_t)b * Ssrc + t) * d;
  float* dp = dst + (size_t)r * d;
  for (int c0 = threadIdx.x; c0 < d; c0 += blockDim.x) dp[c0] = sp[c0];
}
__global__ void k_scatter_rows(const float* __restrict__ src, const int* __restrict__ idx,
                               float* __restrict__ dst, int Sdst, int Ksel, int d) {
  int r = blockIdx.x;
  int b = r / Ksel;
  int t = idx[r];
  const float* sp = src + (size_t)r * d;
  float* dp = dst + ((size_t)b * Sdst + t) * d;
  for (int c0 = threadIdx.x; c0 < d; c0 += blockDim.x) dp[c0] = sp[c0];
}
__global__ void k_zero(float* __restrict__ p, long n) {
  long i = (long)blockIdx.x * blockDim.x + threadIdx.x;
  long st = (long)gridDim.x * blockDim.x;
  for (; i < n; i += st) p[i] = 0.f;
}
__global__ void k_pos(const float* __restrict__ pw, const float* __restrict__ pb,
                      float* __restrict__ posT, int idim, int os_) {
  int o = blockIdx.x;  // posT[o,i] = pos_w[i,o] + pos_b[i]
  for (int i = threadIdx.x; i < idim; i += blockDim.x)
    posT[(size_t)o * idim + i] = pw[(size_t)i * os_ + o] + pb[i];
}
__global__ void k_tile_rows(const float* __restrict__ src, float* __restrict__ dst, int S, int d) {
  int r = blockIdx.x;
  int srow = r % S;
  const float* sp = src + (size_t)srow * d;
  float* dp = dst + (size_t)r * d;
  for (int c0 = threadIdx.x; c0 < d; c0 += blockDim.x) dp[c0] = sp[c0];
}
__global__ void k_bcast_add(const float* __restrict__ bvec, const float* __restrict__ res,
                            float* __restrict__ y, int S, int d) {
  int r = blockIdx.x;
  int srow = r % S;
  const float* bp = bvec + (size_t)srow * d;
  const float* rp = res + (size_t)r * d;
  float* yp = y + (size_t)r * d;
  for (int c0 = threadIdx.x; c0 < d; c0 += blockDim.x) yp[c0] = bp[c0] + rp[c0];
}

// ================= host-side drivers =================
struct BlockP { const float *ln1, *ln2, *wfc, *wo, *wpr, *wqkv; };
struct CombP  { const float *ln2, *lnkv, *lnq, *wfc, *wk, *wo, *wpr, *wq, *wv; };
struct PatchP {
  const float *dec_b, *dec_w;
  BlockP down_scan;
  const float *dp_b, *dp_w;
  CombP ds, ds2;
  const float *enc_b, *enc_w, *gate_b1, *gate_b2, *gate_w1, *gate_w2, *pos_b, *pos_w;
  BlockP query_block;
  const float *up_b, *up_norm;
  BlockP up_scan;
  const float* up_w;
};

static inline void gemm(hipStream_t st, const float* A, long lda, const float* W,
                        const float* bias, const float* res, float* Y,
                        int M, int N, int K, int act) {
  dim3 g((N + 63) / 64, (M + 31) / 32);
  k_gemm<<<g, 128, 0, st>>>(A, lda, W, bias, res, Y, M, N, K, act);
}
static inline void lnorm(hipStream_t st, const float* x, const float* g, float* y, int M, int d) {
  k_ln<<<dim3((M + 3) / 4), 128, 0, st>>>(x, g, y, M, d);
}
static inline void attn(hipStream_t st, const float* q, int ldq, const float* k, int ldk,
                        const float* v, int ldv, float* o, int ldo, int Bn, int S, int d, int causal) {
  int hd = d / NHc;
  dim3 g(S / 16, NHc, Bn);
  if (hd == 16)      k_attn<16><<<g, 32, 0, st>>>(q, ldq, k, ldk, v, ldv, o, ldo, S, causal);
  else if (hd == 32) k_attn<32><<<g, 32, 0, st>>>(q, ldq, k, ldk, v, ldv, o, ldo, S, causal);
  else               k_attn<64><<<g, 32, 0, st>>>(q, ldq, k, ldk, v, ldv, o, ldo, S, causal);
}

static void run_block(hipStream_t st, const float* xin, float* xout, const BlockP& p,
                      int Bn, int S, int d, int causal, float* s1, float* s2, float* qkv, float* f) {
  int M = Bn * S;
  lnorm(st, xin, p.ln1, s1, M, d);
  gemm(st, s1, d, p.wqkv, nullptr, nullptr, qkv, M, 3 * d, d, 0);
  attn(st, qkv, 3 * d, qkv + d, 3 * d, qkv + 2 * d, 3 * d, s1, d, Bn, S, d, causal);
  gemm(st, s1, d, p.wo, nullptr, xin, s2, M, d, d, 0);      // s2 = x + attn@wo^T
  lnorm(st, s2, p.ln2, s1, M, d);
  gemm(st, s1, d, p.wfc, nullptr, nullptr, f, M, 4 * d, d, 1);  // gelu
  gemm(st, f, 4 * d, p.wpr, nullptr, s2, xout, M, d, 4 * d, 0);
}

static void run_combine(hipStream_t st, const float* kv, const float* q, float* out, const CombP& p,
                        int Bn, int S, int d, int causal,
                        float* s1, float* s2, float* s3, float* s4, float* s5, float* f) {
  int M = Bn * S;
  lnorm(st, kv, p.lnkv, s1, M, d);
  lnorm(st, q, p.lnq, s2, M, d);
  gemm(st, s2, d, p.wq, nullptr, nullptr, s3, M, d, d, 0);
  gemm(st, s1, d, p.wk, nullptr, nullptr, s2, M, d, d, 0);
  gemm(st, s1, d, p.wv, nullptr, nullptr, s4, M, d, d, 0);
  attn(st, s3, d, s2, d, s4, d, s1, d, Bn, S, d, causal);
  gemm(st, s1, d, p.wo, nullptr, q, s5, M, d, d, 0);        // s5 = q + attn@wo^T
  lnorm(st, s5, p.ln2, s1, M, d);
  gemm(st, s1, d, p.wfc, nullptr, nullptr, f, M, 4 * d, d, 1);
  gemm(st, f, 4 * d, p.wpr, nullptr, s5, out, M, d, 4 * d, 0);
}

extern "C" void kernel_launch(void* const* d_in, const int* in_sizes, int n_in,
                              void* d_out, int out_size, void* d_ws, size_t ws_size,
                              hipStream_t stream) {
  (void)in_sizes; (void)n_in; (void)out_size; (void)ws_size;
  hipStream_t st = stream;

  // ---- decode inputs (jax tree_leaves order: dict keys sorted at every level) ----
  int cur = 0;
  auto nf = [&]() { return (const float*)d_in[cur++]; };
  auto loadBlock = [&]() { BlockP p; p.ln1 = nf(); p.ln2 = nf(); p.wfc = nf(); p.wo = nf(); p.wpr = nf(); p.wqkv = nf(); return p; };
  auto loadComb  = [&]() { CombP p; p.ln2 = nf(); p.lnkv = nf(); p.lnq = nf(); p.wfc = nf(); p.wk = nf(); p.wo = nf(); p.wpr = nf(); p.wq = nf(); p.wv = nf(); return p; };
  auto loadPatch = [&]() {
    PatchP p;
    p.dec_b = nf(); p.dec_w = nf();
    p.down_scan = loadBlock();
    p.dp_b = nf(); p.dp_w = nf();
    p.ds = loadComb(); p.ds2 = loadComb();
    p.enc_b = nf(); p.enc_w = nf();
    p.gate_b1 = nf(); p.gate_b2 = nf(); p.gate_w1 = nf(); p.gate_w2 = nf();
    p.pos_b = nf(); p.pos_w = nf();
    p.query_block = loadBlock();
    p.up_b = nf(); p.up_norm = nf();
    p.up_scan = loadBlock();
    p.up_w = nf();
    return p;
  };
  const int* tok_idx = (const int*)d_in[cur++];  // "idx"
  const float* head_b = nf();
  const float* head_w = nf();
  BlockP innerb = loadBlock();
  BlockP innerb2 = loadBlock();
  PatchP l0 = loadPatch();
  PatchP l1 = loadPatch();
  const float* pos_emb = nf();
  const float* tok_emb = nf();

  // ---- workspace arena (deterministic bump allocator) ----
  float* base = (float*)d_ws;
  size_t off = 0;
  auto alloc = [&](size_t n) { float* p = base + off; off += (n + 255) & ~(size_t)255; return p; };
  const size_t XSLOT = (size_t)Bc * 256 * 128;  // largest M*d activation
  float* X0 = alloc((size_t)Bc * Tc * D1c);     // residual for l0 down
  float* X1 = alloc((size_t)Bc * K1c * 128);    // residual for l1 down
  float* X2 = alloc((size_t)Bc * K2c * 256);
  float* X3 = alloc((size_t)Bc * K2c * 256);
  float* Y1 = alloc((size_t)Bc * K1c * 128);
  float* Y0 = alloc((size_t)Bc * Tc * D1c);
  float* GATE = alloc((size_t)Bc * Tc);
  float* POS = alloc((size_t)256 * 256);
  float* PDP = alloc((size_t)256 * 128);
  int* IDX0 = (int*)alloc((size_t)Bc * K1c);
  int* IDX1 = (int*)alloc((size_t)Bc * K2c);
  float* SA = alloc(XSLOT); float* SB = alloc(XSLOT); float* SC = alloc(XSLOT); float* SD = alloc(XSLOT);
  float* SE = alloc(XSLOT); float* SF = alloc(XSLOT); float* SG = alloc(XSLOT); float* SH = alloc(XSLOT);
  float* QKV = alloc((size_t)Bc * 256 * 192);
  float* F = alloc((size_t)Bc * 256 * 512);

  // ---- embed ----
  k_embed<<<dim3(Bc * Tc), 64, 0, st>>>(tok_idx, tok_emb, pos_emb, X0, Tc, D1c);

  // ---- level-0 abstract_up (enc/dec autoencoder path is dead code: skipped) ----
  {
    int S = Tc, od = D1c, idim = 128, Ks = K1c, M = Bc * S;
    run_block(st, X0, SA, l0.up_scan, Bc, S, od, 0, SB, SC, QKV, F);
    lnorm(st, SA, l0.up_norm, SB, M, od);
    k_gate<<<dim3(M), 32, 0, st>>>(SB, l0.gate_w1, l0.gate_b1, l0.gate_w2, l0.gate_b2, GATE, od, od / 4);
    run_block(st, X0, SA, l0.query_block, Bc, S, od, 1, SB, SC, QKV, F);
    k_topk<<<dim3(Bc), 32, 0, st>>>(GATE, IDX0, S, Ks);
    k_gather_rows<<<dim3(Bc * Ks), 64, 0, st>>>(SA, IDX0, SB, S, Ks, od);
    gemm(st, SB, od, l0.up_w, l0.up_b, nullptr, X1, Bc * Ks, idim, od, 0);
  }
  // ---- level-1 abstract_up ----
  {
    int S = K1c, od = 128, idim = 256, Ks = K2c, M = Bc * S;
    run_block(st, X1, SA, l1.up_scan, Bc, S, od, 0, SB, SC, QKV, F);
    lnorm(st, SA, l1.up_norm, SB, M, od);
    k_gate<<<dim3(M), 32, 0, st>>>(SB, l1.gate_w1, l1.gate_b1, l1.gate_w2, l1.gate_b2, GATE, od, od / 4);
    run_block(st, X1, SA, l1.query_block, Bc, S, od, 1, SB, SC, QKV, F);
    k_topk<<<dim3(Bc), 32, 0, st>>>(GATE, IDX1, S, Ks);
    k_gather_rows<<<dim3(Bc * Ks), 128, 0, st>>>(SA, IDX1, SB, S, Ks, od);
    gemm(st, SB, od, l1.up_w, l1.up_b, nullptr, X2, Bc * Ks, idim, od, 0);
  }
  // ---- inner blocks (d=256, S=16) ----
  run_block(st, X2, X3, innerb, Bc, K2c, 256, 1, SA, SB, QKV, F);
  run_block(st, X3, X2, innerb2, Bc, K2c, 256, 1, SA, SB, QKV, F);

  // ---- level-1 abstract_down ----
  {
    int Sk = K2c, os_ = K1c, idim = 256, od = 128, M = Bc * os_;
    k_pos<<<dim3(os_), 128, 0, st>>>(l1.pos_w, l1.pos_b, POS, idim, os_);
    k_zero<<<dim3(1024), 256, 0, st>>>(SA, (long)M * idim);
    k_scatter_rows<<<dim3(Bc * Sk), 128, 0, st>>>(X2, IDX1, SA, os_, Sk, idim);
    k_tile_rows<<<dim3(M), 128, 0, st>>>(POS, SB, os_, idim);
    run_combine(st, SA, SB, SC, l1.ds, Bc, os_, idim, 1, SD, SE, SF, SG, SH, F);
    gemm(st, SC, idim, l1.dp_w, l1.dp_b, nullptr, SD, M, od, idim, 0);   // pds
    gemm(st, POS, idim, l1.dp_w, l1.dp_b, nullptr, PDP, os_, od, idim, 0);
    k_bcast_add<<<dim3(M), 128, 0, st>>>(PDP, X1, SE, os_, od);          // query
    run_combine(st, SD, SE, SA, l1.ds2, Bc, os_, od, 1, SB, SC, SF, SG, SH, F);
    run_block(st, SA, Y1, l1.down_scan, Bc, os_, od, 1, SB, SC, QKV, F);
  }
  // ---- level-0 abstract_down ----
  {
    int Sk = K1c, os_ = Tc, idim = 128, od = D1c, M = Bc * os_;
    k_pos<<<dim3(os_), 128, 0, st>>>(l0.pos_w, l0.pos_b, POS, idim, os_);
    k_zero<<<dim3(2048), 256, 0, st>>>(SA, (long)M * idim);
    k_scatter_rows<<<dim3(Bc * Sk), 128, 0, st>>>(Y1, IDX0, SA, os_, Sk, idim);
    k_tile_rows<<<dim3(M), 128, 0, st>>>(POS, SB, os_, idim);
    run_combine(st, SA, SB, SC, l0.ds, Bc, os_, idim, 1, SD, SE, SF, SG, SH, F);
    gemm(st, SC, idim, l0.dp_w, l0.dp_b, nullptr, SD, M, od, idim, 0);   // pds
    gemm(st, POS, idim, l0.dp_w, l0.dp_b, nullptr, PDP, os_, od, idim, 0);
    k_bcast_add<<<dim3(M), 128, 0, st>>>(PDP, X0, SE, os_, od);          // query
    run_combine(st, SD, SE, SA, l0.ds2, Bc, os_, od, 1, SB, SC, SF, SG, SH, F);
    run_block(st, SA, Y0, l0.down_scan, Bc, os_, od, 1, SB, SC, QKV, F);
  }
  // ---- vocab head on last token: logits[b] = Y0[b, T-1, :] @ head_w^T + head_b ----
  gemm(st, Y0 + (size_t)(Tc - 1) * D1c, (long)Tc * D1c, head_w, head_b, nullptr,
       (float*)d_out, Bc, Vc, D1c, 0);
}
